// EnsembleSelfAttention_5600637354334
// MI455X (gfx1250) — compile-verified, matrix operand order checked by reading, not run on hardware
//
#include <hip/hip_runtime.h>
#include <hip/hip_bf16.h>

// ---------------------------------------------------------------------------
// EnsembleSelfAttention on gfx1250 (MI455X), bf16 WMMA pipeline.
// B=8, E=50, C=64, H=32, W=64 (HW=2048), NH=64
// ---------------------------------------------------------------------------

#define BB   8
#define EE   50
#define CCH  64
#define HWN  2048
#define NHN  64
#define BE   (BB * EE)          // 400
#define BC   (BB * NHN)         // 512
#define GRP  (CCH * HWN)        // 131072 elements per (b,e) LayerNorm group

__device__ __forceinline__ float scale_const() { return 0.022097086912079608f; } // (32*64)^-0.5

typedef __attribute__((ext_vector_type(16))) __bf16 v16bf;
typedef __attribute__((ext_vector_type(8)))  float  v8f;
typedef __attribute__((ext_vector_type(2)))  __bf16 bf16x2;

union BF16Vec {
  v16bf v;
  unsigned int u[8];
};

// Native f32->bf16 (RNE); backend can fuse into v_cvt_pk_bf16_f32.
__device__ __forceinline__ unsigned int pack2bf(float lo, float hi) {
  bf16x2 t;
  t[0] = (__bf16)lo;
  t[1] = (__bf16)hi;
  return __builtin_bit_cast(unsigned int, t);
}
__device__ __forceinline__ unsigned short bfbits(float f) {
  return __builtin_bit_cast(unsigned short, (__bf16)f);
}

// CDNA5 16-bit A/B-operand K placement: VGPR vg holds K pair starting at:
//   lanes 0-15 : vg<4 -> 2*vg        ; vg>=4 -> 16 + 2*(vg-4)
//   lanes 16-31: same + 8
__device__ __forceinline__ int kpair_base(int vg, int lane) {
  int kh = (lane & 16) ? 8 : 0;
  return (vg < 4) ? (kh + 2 * vg) : (kh + 16 + 2 * (vg - 4));
}

__device__ __forceinline__ v8f bwmma(BF16Vec a, BF16Vec b, v8f c) {
  return __builtin_amdgcn_wmma_f32_16x16x32_bf16(false, a.v, false, b.v,
                                                 (short)0, c, false, false);
}

// ---------------------------------------------------------------------------
// Kernel 0: pre-swizzle Wq/Wk/Wv/Wo into WMMA B-operand layout (bf16).
// Group g = widx*4 + nt (widx: 0=Wq,1=Wk,2=Wv,3=Wo). Per lane 16 uints:
// [0..7] = K 0..31 operand, [8..15] = K 32..63 operand.
// ---------------------------------------------------------------------------
__global__ __launch_bounds__(32) void k_prepw(const float* __restrict__ Wq,
                                              const float* __restrict__ Wk,
                                              const float* __restrict__ Wv,
                                              const float* __restrict__ Wo,
                                              unsigned int* __restrict__ wsw) {
  const int g = blockIdx.x;                        // 0..15
  const int widx = g >> 2, nt = g & 3;
  const float* W = (widx == 0) ? Wq : (widx == 1) ? Wk : (widx == 2) ? Wv : Wo;
  const int lane = threadIdx.x & 31;
  const int ncol = nt * 16 + (lane & 15);
  unsigned int* dst = wsw + ((size_t)g * 32 + lane) * 16;
#pragma unroll
  for (int vg = 0; vg < 8; ++vg) {
    int k0 = kpair_base(vg, lane);
    dst[vg]     = pack2bf(W[k0 * 64 + ncol],        W[(k0 + 1) * 64 + ncol]);
    dst[vg + 8] = pack2bf(W[(k0 + 32) * 64 + ncol], W[(k0 + 33) * 64 + ncol]);
  }
}

__device__ __forceinline__ void load_wtile(const unsigned int* __restrict__ wsw,
                                           int g, int lane, BF16Vec& b0, BF16Vec& b1) {
  const uint4* wp = reinterpret_cast<const uint4*>(wsw + ((size_t)g * 32 + lane) * 16);
  uint4 w0 = wp[0], w1 = wp[1], w2 = wp[2], w3 = wp[3];
  b0.u[0] = w0.x; b0.u[1] = w0.y; b0.u[2] = w0.z; b0.u[3] = w0.w;
  b0.u[4] = w1.x; b0.u[5] = w1.y; b0.u[6] = w1.z; b0.u[7] = w1.w;
  b1.u[0] = w2.x; b1.u[1] = w2.y; b1.u[2] = w2.z; b1.u[3] = w2.w;
  b1.u[4] = w3.x; b1.u[5] = w3.y; b1.u[6] = w3.z; b1.u[7] = w3.w;
}

// ---------------------------------------------------------------------------
// Kernel 1: LayerNorm statistics per (b,e): mean and rstd.
// ---------------------------------------------------------------------------
__global__ __launch_bounds__(256) void k_stats(const float* __restrict__ x,
                                               float* __restrict__ stats) {
  const int be = blockIdx.x;                       // 0..399
  const float* p = x + (size_t)be * GRP;
  float s = 0.f, q = 0.f;
  for (int i = threadIdx.x; i < GRP; i += 256) {
    float v = p[i];
    s += v;
    q += v * v;
  }
  __shared__ float ss[256];
  __shared__ float sq[256];
  ss[threadIdx.x] = s;
  sq[threadIdx.x] = q;
  __syncthreads();
  for (int st = 128; st > 0; st >>= 1) {
    if (threadIdx.x < st) {
      ss[threadIdx.x] += ss[threadIdx.x + st];
      sq[threadIdx.x] += sq[threadIdx.x + st];
    }
    __syncthreads();
  }
  if (threadIdx.x == 0) {
    float mean = ss[0] * (1.0f / (float)GRP);
    float var  = sq[0] * (1.0f / (float)GRP) - mean * mean;
    stats[2 * be]     = mean;
    stats[2 * be + 1] = rsqrtf(var + 1e-5f);
  }
}

// ---------------------------------------------------------------------------
// Kernel 2: fused normalize + q/k/v projection.
// grid = (16, 400), block = 256 (8 waves). Each wave: one 16(hw)x64(c) A tile,
// 3 weights x 4 n-tiles x 2 WMMAs. Output bf16 [b][nh][e][hw].
// ---------------------------------------------------------------------------
__global__ __launch_bounds__(256) void k_qkv(const float* __restrict__ x,
                                             const float* __restrict__ lnw,
                                             const float* __restrict__ lnb,
                                             const float* __restrict__ stats,
                                             const unsigned int* __restrict__ wsw,
                                             unsigned short* __restrict__ qb,
                                             unsigned short* __restrict__ kb,
                                             unsigned short* __restrict__ vb) {
  const int be   = blockIdx.y;
  const int b    = be / EE, e = be % EE;
  const int wid  = threadIdx.x >> 5;
  const int lane = threadIdx.x & 31;
  const int l15  = lane & 15;
  const int hi8  = (lane & 16) ? 8 : 0;
  const int hwb  = (blockIdx.x * 8 + wid) * 16;
  const int hw   = hwb + l15;                      // A row for this lane
  const float mean = stats[2 * be];
  const float rstd = stats[2 * be + 1];
  const float* xp = x + (size_t)be * GRP;

  // ---- A operands: n[hw, c] for c=0..31 (a0) and 32..63 (a1).
  // Phase 1: bulk loads into register arrays (no conversion consumers yet).
  float xr[32], wr[32], br[32];
#pragma unroll
  for (int vg = 0; vg < 8; ++vg) {
    const int k0 = kpair_base(vg, lane);
    const size_t o0 = (size_t)k0 * HWN + hw;
    xr[2 * vg]      = xp[o0];
    xr[2 * vg + 1]  = xp[o0 + HWN];
    xr[16 + 2 * vg]     = xp[o0 + 32 * HWN];
    xr[16 + 2 * vg + 1] = xp[o0 + 33 * HWN];
    wr[2 * vg]      = lnw[o0];
    wr[2 * vg + 1]  = lnw[o0 + HWN];
    wr[16 + 2 * vg]     = lnw[o0 + 32 * HWN];
    wr[16 + 2 * vg + 1] = lnw[o0 + 33 * HWN];
    br[2 * vg]      = lnb[o0];
    br[2 * vg + 1]  = lnb[o0 + HWN];
    br[16 + 2 * vg]     = lnb[o0 + 32 * HWN];
    br[16 + 2 * vg + 1] = lnb[o0 + 33 * HWN];
  }
  // Phase 2: normalize + convert + pack.
  BF16Vec a0, a1;
#pragma unroll
  for (int vg = 0; vg < 8; ++vg) {
    float n00 = (xr[2 * vg] - mean) * rstd * wr[2 * vg] + br[2 * vg];
    float n01 = (xr[2 * vg + 1] - mean) * rstd * wr[2 * vg + 1] + br[2 * vg + 1];
    float n10 = (xr[16 + 2 * vg] - mean) * rstd * wr[16 + 2 * vg] + br[16 + 2 * vg];
    float n11 = (xr[16 + 2 * vg + 1] - mean) * rstd * wr[16 + 2 * vg + 1] + br[16 + 2 * vg + 1];
    a0.u[vg] = pack2bf(n00, n01);
    a1.u[vg] = pack2bf(n10, n11);
  }

  unsigned short* dst[3] = {qb, kb, vb};
#pragma unroll
  for (int ws = 0; ws < 3; ++ws) {
#pragma unroll
    for (int nt = 0; nt < 4; ++nt) {
      const int ncol = nt * 16 + l15;              // B column (= nh) per lane
      BF16Vec b0, b1;
      load_wtile(wsw, ws * 4 + nt, lane, b0, b1);
      v8f acc = {};
      acc = bwmma(a0, b0, acc);
      acc = bwmma(a1, b1, acc);
      // D element (m,n): hw = hwb + m, nh = ncol.  Per lane: 8 consecutive hw
      // -> one packed 16B store to [b][nh][e][hw].
      uint4 val;
      val.x = pack2bf(acc[0], acc[1]);
      val.y = pack2bf(acc[2], acc[3]);
      val.z = pack2bf(acc[4], acc[5]);
      val.w = pack2bf(acc[6], acc[7]);
      size_t off = (((size_t)(b * NHN + ncol) * EE + e) * HWN) + hwb + hi8;
      *reinterpret_cast<uint4*>(dst[ws] + off) = val;
    }
  }
}

// ---------------------------------------------------------------------------
// Kernel 3: dot[b,nh,j,i] = sum_hw k[i,hw] * q[j,hw]   (K = 2048, bf16 WMMA)
// grid = 512 (b*nh), block = 256; each wave handles 2 of the 16 (i,j) tiles.
// ---------------------------------------------------------------------------
__global__ __launch_bounds__(256) void k_dot(const unsigned short* __restrict__ kb,
                                             const unsigned short* __restrict__ qb,
                                             float* __restrict__ dot) {
  const int bc   = blockIdx.x;                     // b*64 + nh
  const int wid  = threadIdx.x >> 5;
  const int lane = threadIdx.x & 31;
  const int l15  = lane & 15;
  const int hi8  = (lane & 16) ? 8 : 0;

  for (int tt = 0; tt < 2; ++tt) {
    const int t    = wid + tt * 8;                 // 0..15
    const int it   = t >> 2, jt = t & 3;
    const int irow = it * 16 + l15;                // A row (ensemble i)
    const int jcol = jt * 16 + l15;                // B col (ensemble j)
    const bool iok = irow < EE;
    const bool jok = jcol < EE;
    const size_t kbase = ((size_t)bc * EE + irow) * HWN;
    const size_t qbase = ((size_t)bc * EE + jcol) * HWN;

    v8f acc = {};
#pragma unroll 2
    for (int kk = 0; kk < 64; ++kk) {
      const int hwb = kk * 32;
      BF16Vec a, bvec;
#pragma unroll
      for (int vg = 0; vg < 8; ++vg) {
        int k0 = kpair_base(vg, lane);             // K pairs are contiguous hw
        a.u[vg]    = iok ? *reinterpret_cast<const unsigned int*>(kb + kbase + hwb + k0) : 0u;
        bvec.u[vg] = jok ? *reinterpret_cast<const unsigned int*>(qb + qbase + hwb + k0) : 0u;
      }
      acc = bwmma(a, bvec, acc);
    }
    // D rows i contiguous per lane -> two float4 stores to [b][nh][j][i]
    size_t off = ((size_t)bc * 64 + jcol) * 64 + it * 16 + hi8;
    float4 v0 = make_float4(acc[0], acc[1], acc[2], acc[3]);
    float4 v1 = make_float4(acc[4], acc[5], acc[6], acc[7]);
    *reinterpret_cast<float4*>(dot + off)     = v0;
    *reinterpret_cast<float4*>(dot + off + 4) = v1;
  }
}

// ---------------------------------------------------------------------------
// Kernel 4: softmax over i (SCALE folded in); write bf16 attn[b][nh][j][i],
// zero-padded to i,j in [0,64) so downstream WMMA tiles are clean.
// ---------------------------------------------------------------------------
__global__ __launch_bounds__(64) void k_softmax(const float* __restrict__ dot,
                                                unsigned short* __restrict__ attn) {
  const int bc = blockIdx.x;                       // 512
  const int j  = threadIdx.x;                      // 0..63
  const float* row = dot + ((size_t)bc * 64 + j) * 64;
  unsigned short* arow = attn + ((size_t)bc * 64 + j) * 64;
  if (j >= EE) {
    for (int i = 0; i < 64; ++i) arow[i] = 0;
    return;
  }
  const float sc = scale_const();
  float mx = -3.0e38f;
  for (int i = 0; i < EE; ++i) mx = fmaxf(mx, row[i] * sc);
  float s = 0.f;
  for (int i = 0; i < EE; ++i) s += __expf(row[i] * sc - mx);
  float inv = 1.f / s;
  for (int i = 0; i < EE; ++i) arow[i] = bfbits(__expf(row[i] * sc - mx) * inv);
  for (int i = EE; i < 64; ++i) arow[i] = 0;
}

// ---------------------------------------------------------------------------
// Kernel 5: tv[j,hw] = sum_i attn[i,j] * v[i,hw]  (K = 64 padded, bf16 WMMA)
// grid = 512 (b*nh), block = 256; each wave: one j-tile x 64 hw-tiles.
// Output bf16 tv[b][e][hw][nh] (contiguous-nh for the final projection).
// ---------------------------------------------------------------------------
__global__ __launch_bounds__(256) void k_tv(const unsigned short* __restrict__ attn,
                                            const unsigned short* __restrict__ vb,
                                            unsigned short* __restrict__ tv) {
  const int bc   = blockIdx.x;
  const int b    = bc >> 6, nh = bc & 63;
  const int wid  = threadIdx.x >> 5;
  const int lane = threadIdx.x & 31;
  const int l15  = lane & 15;
  const int hi8  = (lane & 16) ? 8 : 0;
  const int jt   = wid >> 1;
  const int jrow = jt * 16 + l15;

  // A = attn^T[j, i]; K pairs contiguous in i.
  BF16Vec a0, a1;
  const size_t abase = ((size_t)bc * 64 + jrow) * 64;
#pragma unroll
  for (int vg = 0; vg < 8; ++vg) {
    int k0 = kpair_base(vg, lane);
    a0.u[vg] = *reinterpret_cast<const unsigned int*>(attn + abase + k0);
    a1.u[vg] = *reinterpret_cast<const unsigned int*>(attn + abase + 32 + k0);
  }

  const size_t vrow = (size_t)bc * EE;             // v[b][nh][i][hw]
  for (int h = 0; h < 64; ++h) {
    const int hwt = (wid & 1) * 64 + h;
    const int hw  = hwt * 16 + l15;                // B column (hw)
    BF16Vec b0, b1;
#pragma unroll
    for (int vg = 0; vg < 8; ++vg) {
      int k0 = kpair_base(vg, lane);
      // K dimension is i; stride between i's is HWN, so element loads.
      unsigned int lo = vb[(vrow + k0) * HWN + hw];          // k0   <= 31 < 50
      unsigned int hi = vb[(vrow + k0 + 1) * HWN + hw];      // k0+1 <= 32 < 50
      b0.u[vg] = lo | (hi << 16);
      int k1 = k0 + 32;
      lo = (k1 < EE)     ? (unsigned int)vb[(vrow + k1) * HWN + hw]     : 0u;
      hi = (k1 + 1 < EE) ? (unsigned int)vb[(vrow + k1 + 1) * HWN + hw] : 0u;
      b1.u[vg] = lo | (hi << 16);
    }
    v8f acc = {};
    acc = bwmma(a0, b0, acc);
    acc = bwmma(a1, b1, acc);
    // D rows j (guard j<50), cols hw. Store to tv[b][e][hw][nh].
#pragma unroll
    for (int r = 0; r < 8; ++r) {
      int j = jt * 16 + hi8 + r;
      if (j < EE)
        tv[(((size_t)(b * EE + j)) * HWN + hw) * NHN + nh] = bfbits(acc[r]);
    }
  }
}

// ---------------------------------------------------------------------------
// Kernel 6: out[b,e,c,hw] = x + tv @ Wo  (K = 64, bf16 WMMA + fp32 residual)
// grid = (16, 400), block = 256.
// ---------------------------------------------------------------------------
__global__ __launch_bounds__(256) void k_out(const float* __restrict__ x,
                                             const unsigned short* __restrict__ tv,
                                             const unsigned int* __restrict__ wsw,
                                             float* __restrict__ out) {
  const int be   = blockIdx.y;
  const int wid  = threadIdx.x >> 5;
  const int lane = threadIdx.x & 31;
  const int l15  = lane & 15;
  const int hi8  = (lane & 16) ? 8 : 0;
  const int hwb  = (blockIdx.x * 8 + wid) * 16;

  // A = tv[(e,hw) row, nh]; K pairs contiguous in nh.
  BF16Vec a0, a1;
  const size_t abase = (((size_t)be * HWN) + hwb + l15) * NHN;
#pragma unroll
  for (int vg = 0; vg < 8; ++vg) {
    int k0 = kpair_base(vg, lane);
    a0.u[vg] = *reinterpret_cast<const unsigned int*>(tv + abase + k0);
    a1.u[vg] = *reinterpret_cast<const unsigned int*>(tv + abase + 32 + k0);
  }

#pragma unroll
  for (int ct = 0; ct < 4; ++ct) {
    const int c = ct * 16 + l15;                   // B column (output channel)
    BF16Vec b0, b1;
    load_wtile(wsw, 12 + ct, lane, b0, b1);        // Wo tiles
    v8f acc = {};
    acc = bwmma(a0, b0, acc);
    acc = bwmma(a1, b1, acc);
    // Per lane: 8 consecutive hw at fixed (b,e,c) -> coalesced float4 IO.
    size_t off = (((size_t)be * CCH) + c) * HWN + hwb + hi8;
    float4 x0 = *reinterpret_cast<const float4*>(x + off);
    float4 x1 = *reinterpret_cast<const float4*>(x + off + 4);
    float4 o0 = make_float4(x0.x + acc[0], x0.y + acc[1], x0.z + acc[2], x0.w + acc[3]);
    float4 o1 = make_float4(x1.x + acc[4], x1.y + acc[5], x1.z + acc[6], x1.w + acc[7]);
    *reinterpret_cast<float4*>(out + off)     = o0;
    *reinterpret_cast<float4*>(out + off + 4) = o1;
  }
}

// ---------------------------------------------------------------------------
// Launch
// ---------------------------------------------------------------------------
extern "C" void kernel_launch(void* const* d_in, const int* in_sizes, int n_in,
                              void* d_out, int out_size, void* d_ws, size_t ws_size,
                              hipStream_t stream) {
  (void)in_sizes; (void)n_in; (void)out_size; (void)ws_size;
  const float* x   = (const float*)d_in[0];
  const float* lnw = (const float*)d_in[1];
  const float* lnb = (const float*)d_in[2];
  const float* Wq  = (const float*)d_in[3];
  const float* Wk  = (const float*)d_in[4];
  const float* Wv  = (const float*)d_in[5];
  const float* Wo  = (const float*)d_in[6];
  float* out = (float*)d_out;

  char* ws = (char*)d_ws;
  const size_t QK_BYTES   = (size_t)BB * NHN * EE * HWN * 2;  // 104,857,600
  const size_t DOT_BYTES  = (size_t)BC * 64 * 64 * 4;         //   8,388,608
  const size_t ATTN_BYTES = (size_t)BC * 64 * 64 * 2;         //   4,194,304
  float*          stats = (float*)(ws);
  unsigned short* qb    = (unsigned short*)(ws + 4096);
  unsigned short* kb    = (unsigned short*)(ws + 4096 + QK_BYTES);
  unsigned short* vb    = (unsigned short*)(ws + 4096 + 2 * QK_BYTES);
  float*          dotb  = (float*)(ws + 4096 + 3 * QK_BYTES);
  unsigned short* attnb = (unsigned short*)(ws + 4096 + 3 * QK_BYTES + DOT_BYTES);
  unsigned int*   wsw   = (unsigned int*)(ws + 4096 + 3 * QK_BYTES + DOT_BYTES + ATTN_BYTES);
  unsigned short* tvb   = qb;   // q is dead after k_dot; alias tv onto it

  k_prepw  <<<16, 32, 0, stream>>>(Wq, Wk, Wv, Wo, wsw);
  k_stats  <<<BE, 256, 0, stream>>>(x, stats);
  k_qkv    <<<dim3(16, BE), 256, 0, stream>>>(x, lnw, lnb, stats, wsw, qb, kb, vb);
  k_dot    <<<BC, 256, 0, stream>>>(kb, qb, dotb);
  k_softmax<<<BC, 64, 0, stream>>>(dotb, attnb);
  k_tv     <<<BC, 256, 0, stream>>>(attnb, vb, tvb);
  k_out    <<<dim3(16, BE), 256, 0, stream>>>(x, tvb, wsw, out);
}
